// Tokenizer_26250840113297
// MI455X (gfx1250) — matrix operand order repaired and checked by reading, not compiled, not run on hardware
//
#include <hip/hip_runtime.h>
#include <hip/hip_fp16.h>

typedef _Float16 v16h __attribute__((ext_vector_type(16)));
typedef _Float16 v8h  __attribute__((ext_vector_type(8)));
typedef float    v8f  __attribute__((ext_vector_type(8)));

#define CDIM 256
#define VCODES 511
#define NPAD 512
#define EH_STRIDE 264          // halves per LDS e-row (528B, bank-conflict padding)
#define M_BLOCK 64             // 4 row-groups of 16, two waves (N-halves) each
#define NWAVES 8
#define ROWS_TOTAL (32*4096)   // 131072 tokens

// ---------------- prep: e -> f16 in ws, ||e||^2, zero loss accumulators ----
__global__ void k_prep(const float* __restrict__ ew, _Float16* __restrict__ ws_eh,
                       float* __restrict__ ws_enorm, float* __restrict__ ws_loss) {
  const int j = blockIdx.x;      // 0..511 (code index; pad row 511)
  const int tid = threadIdx.x;   // 0..255
  if (j == 0 && tid < 4) ws_loss[tid] = 0.0f;
  __shared__ float red[256];
  float v = 0.0f;
  if (j < VCODES) v = ew[(size_t)(j + 1) * CDIM + tid];   // e = e_weight[1:]
  ws_eh[(size_t)j * CDIM + tid] = (_Float16)v;
  red[tid] = v * v;
  __syncthreads();
  for (int s = 128; s > 0; s >>= 1) {
    if (tid < s) red[tid] += red[tid + s];
    __syncthreads();
  }
  if (tid == 0) ws_enorm[j] = (j < VCODES) ? red[0] : 1e30f;
}

// ---------------- inverse norms of z rows ---------------------------------
__global__ void k_invnorm(const float* __restrict__ z, float* __restrict__ invn) {
  const int w = threadIdx.x >> 5, l = threadIdx.x & 31;
  const int row = blockIdx.x * 8 + w;
  const float4* p = (const float4*)(z + (size_t)row * CDIM) + l * 2;
  float4 a = p[0], b = p[1];
  float s = a.x*a.x + a.y*a.y + a.z*a.z + a.w*a.w
          + b.x*b.x + b.y*b.y + b.z*b.z + b.w*b.w;
#pragma unroll
  for (int m = 1; m < 32; m <<= 1) s += __shfl_xor(s, m, 32);
  if (l == 0) invn[row] = 1.0f / fmaxf(sqrtf(s), 1e-12f);
}

// ---------------- sum(mask) -> ws_loss[2] ---------------------------------
__global__ void k_masksum(const float* __restrict__ mask, float* __restrict__ ws_loss) {
  __shared__ float red[256];
  float s = 0.0f;
  for (int i = blockIdx.x * 256 + threadIdx.x; i < ROWS_TOTAL; i += 64 * 256)
    s += mask[i];
  red[threadIdx.x] = s;
  __syncthreads();
  for (int st = 128; st > 0; st >>= 1) {
    if (threadIdx.x < st) red[threadIdx.x] += red[threadIdx.x + st];
    __syncthreads();
  }
  if (threadIdx.x == 0) atomicAdd(ws_loss + 2, red[0]);
}

// ---------------- main: GEMM (WMMA f16) + log-softmax + commitment --------
// 8 waves = 4 row-groups (16 rows) x 2 N-halves (16 tiles of 16 columns each).
// Each wave holds 16 f32 accumulators (128 VGPRs) -> total regs < 256, no
// vgpr-msb switching. Wave pairs exchange row max / sumexp through LDS.
__global__ void __launch_bounds__(256, 1)
k_main(const float* __restrict__ z, const float* __restrict__ mask,
       const _Float16* __restrict__ ws_eh, const float* __restrict__ ws_enorm,
       const float* __restrict__ invn, const int* __restrict__ stepp,
       float* __restrict__ logp, float* __restrict__ ws_loss) {
  extern __shared__ _Float16 eh[];           // [512][EH_STRIDE]
  __shared__ float redmax[NWAVES][16];
  __shared__ float redsum[NWAVES][16];
  __shared__ float sred[4];
  const int tid = threadIdx.x;

  // stage f16 codebook into LDS (stride-padded), 16B chunks
  for (int c = tid; c < NPAD * 32; c += 256) {
    const int row = c >> 5, off = (c & 31) << 3;
    *(uint4*)(eh + row * EH_STRIDE + off) = *(const uint4*)(ws_eh + row * CDIM + off);
  }
  __syncthreads();

  const int w = tid >> 5, lane = tid & 31;
  const int r = lane & 15, h = lane >> 4;
  const int p = w >> 1;            // row-group 0..3
  const int nh = w & 1;            // which N-half (tiles nh*16 .. nh*16+15)
  const int m0 = blockIdx.x * M_BLOCK + p * 16;
  const int rowA = m0 + r;
  const float stepf = (float)(*stepp);            // TEMP == 1.0
  const float inA = invn[rowA];
  const float* zrow = z + (size_t)rowA * CDIM;

  v8f acc[16];
#pragma unroll
  for (int n = 0; n < 16; n++) acc[n] = v8f{};

#pragma unroll
  for (int kk = 0; kk < 8; kk++) {
    // A fragment (16x32 f16): lane half h -> K = 32kk + 8h + {0..7, 16..23}
    const int base = kk * 32 + h * 8;
    float4 f0 = *(const float4*)(zrow + base);
    float4 f1 = *(const float4*)(zrow + base + 4);
    float4 f2 = *(const float4*)(zrow + base + 16);
    float4 f3 = *(const float4*)(zrow + base + 20);
    v16h a;
    a[0]=(_Float16)(f0.x*inA); a[1]=(_Float16)(f0.y*inA);
    a[2]=(_Float16)(f0.z*inA); a[3]=(_Float16)(f0.w*inA);
    a[4]=(_Float16)(f1.x*inA); a[5]=(_Float16)(f1.y*inA);
    a[6]=(_Float16)(f1.z*inA); a[7]=(_Float16)(f1.w*inA);
    a[8]=(_Float16)(f2.x*inA); a[9]=(_Float16)(f2.y*inA);
    a[10]=(_Float16)(f2.z*inA); a[11]=(_Float16)(f2.w*inA);
    a[12]=(_Float16)(f3.x*inA); a[13]=(_Float16)(f3.y*inA);
    a[14]=(_Float16)(f3.z*inA); a[15]=(_Float16)(f3.w*inA);
#pragma unroll
    for (int nl = 0; nl < 16; nl++) {
      const int tn = nh * 16 + nl;  // global N tile
      // B fragment (32x16 f16): lane r = column (code tn*16+r), K = 32kk + 8h + {0..7, 16..23}
      const _Float16* bp = eh + (tn * 16 + r) * EH_STRIDE + kk * 32 + h * 8;
      v8h b0 = *(const v8h*)bp;
      v8h b1 = *(const v8h*)(bp + 16);
      v16h bfr = __builtin_shufflevector(b0, b1, 0,1,2,3,4,5,6,7,8,9,10,11,12,13,14,15);
      acc[nl] = __builtin_amdgcn_wmma_f32_16x16x32_f16(
          false, a, false, bfr, (short)0, acc[nl], false, false);
    }
  }

  // scores s = -d*step = step*(2*dot - ||e||^2 - 1)   (||zn||^2 == 1)
#pragma unroll
  for (int nl = 0; nl < 16; nl++) {
    const float en = ws_enorm[(nh * 16 + nl) * 16 + r];
#pragma unroll
    for (int g = 0; g < 8; g++)
      acc[nl][g] = stepf * (2.0f * acc[nl][g] - en - 1.0f);
  }

  // per-row max over own half (lane = column; masks 1..8 stay in 16-lane half)
  float rmax[8];
#pragma unroll
  for (int g = 0; g < 8; g++) {
    float m = acc[0][g];
#pragma unroll
    for (int nl = 1; nl < 16; nl++) m = fmaxf(m, acc[nl][g]);
#pragma unroll
    for (int mk = 1; mk < 16; mk <<= 1) m = fmaxf(m, __shfl_xor(m, mk, 32));
    rmax[g] = m;
  }
  // exchange max with partner wave (other N-half, same rows)
  if (r == 0) {
#pragma unroll
    for (int g = 0; g < 8; g++) redmax[w][h * 8 + g] = rmax[g];
  }
  __syncthreads();
  float fullmax[8];
#pragma unroll
  for (int g = 0; g < 8; g++) fullmax[g] = fmaxf(rmax[g], redmax[w ^ 1][h * 8 + g]);

  // partial sum of exp over own half
  float psum[8];
#pragma unroll
  for (int g = 0; g < 8; g++) {
    float s = 0.0f;
#pragma unroll
    for (int nl = 0; nl < 16; nl++) s += __expf(acc[nl][g] - fullmax[g]);
#pragma unroll
    for (int mk = 1; mk < 16; mk <<= 1) s += __shfl_xor(s, mk, 32);
    psum[g] = s;
  }
  if (r == 0) {
#pragma unroll
    for (int g = 0; g < 8; g++) redsum[w][h * 8 + g] = psum[g];
  }
  __syncthreads();
  float lsum[8];
#pragma unroll
  for (int g = 0; g < 8; g++)
    lsum[g] = fullmax[g] + __logf(psum[g] + redsum[w ^ 1][h * 8 + g]);

  // write log_probs (apply mask); nh==0 waves accumulate commitment
  float cpart = 0.0f;
#pragma unroll
  for (int g = 0; g < 8; g++) {
    const int row = m0 + h * 8 + g;
    const float mv = mask[row];
    if (nh == 0) cpart += (-fullmax[g] / stepf) * mv;   // d_min = -s_max/step
    const float corr = lsum[g];
    const float neg = (1.0f - mv) * (-1e9f);
#pragma unroll
    for (int nl = 0; nl < 16; nl++) {
      const int col = (nh * 16 + nl) * 16 + r;
      if (col < VCODES)
        logp[(size_t)row * VCODES + col] = (acc[nl][g] - corr) * mv + neg;
    }
  }
  // cpart identical across each 16-lane half; combine halves, reduce block
  if (nh == 0) {
    const float other = __shfl_xor(cpart, 16, 32);
    if (lane == 0) sred[p] = cpart + other;
  }
  __syncthreads();
  if (tid == 0) {
    float t = 0.0f;
    for (int i = 0; i < 4; i++) t += sred[i];
    atomicAdd(ws_loss + 1, t);
  }
}

// ---------------- smoothness: 2 - 2*zn_t . zn_{t+1} -----------------------
__global__ void k_smooth(const float* __restrict__ z, const float* __restrict__ mask,
                         const float* __restrict__ invn, float* __restrict__ ws_loss) {
  __shared__ float sred[NWAVES];
  const int tid = threadIdx.x, w = tid >> 5, lane = tid & 31;
  float part = 0.0f;
  for (int i = 0; i < 16; i++) {
    const int slot = blockIdx.x * 128 + w * 16 + i;     // 1024 blocks * 128 = 131072
    const int t = slot & 4095;
    if (t == 4095) continue;                            // no pair across batches
    const float4* pa = (const float4*)(z + (size_t)slot * CDIM) + lane * 2;
    const float4* pb = (const float4*)(z + (size_t)(slot + 1) * CDIM) + lane * 2;
    float4 a0 = pa[0], a1 = pa[1], b0 = pb[0], b1 = pb[1];
    float d = a0.x*b0.x + a0.y*b0.y + a0.z*b0.z + a0.w*b0.w
            + a1.x*b1.x + a1.y*b1.y + a1.z*b1.z + a1.w*b1.w;
#pragma unroll
    for (int mk = 1; mk < 32; mk <<= 1) d += __shfl_xor(d, mk, 32);
    if (lane == 0)
      part += mask[slot + 1] * (2.0f - 2.0f * d * invn[slot] * invn[slot + 1]);
  }
  if (lane == 0) sred[w] = part;
  __syncthreads();
  if (tid == 0) {
    float s = 0.0f;
    for (int i = 0; i < NWAVES; i++) s += sred[i];
    atomicAdd(ws_loss + 0, s);
  }
}

// ---------------- finalize scalars ----------------------------------------
__global__ void k_final(const float* __restrict__ ws_loss, float* __restrict__ out) {
  if (threadIdx.x == 0) {
    float vc = fmaxf(ws_loss[2] * (float)CDIM, 1.0f);   // sum(mask) * c
    out[0] = ws_loss[0] / vc;   // smoothness
    out[1] = ws_loss[1] / vc;   // commitment
  }
}

extern "C" void kernel_launch(void* const* d_in, const int* in_sizes, int n_in,
                              void* d_out, int out_size, void* d_ws, size_t ws_size,
                              hipStream_t stream) {
  const float* z    = (const float*)d_in[0];   // [32,4096,256]
  const float* mask = (const float*)d_in[1];   // [32,4096,1]
  const float* ew   = (const float*)d_in[2];   // [512,256]
  const int*   step = (const int*)d_in[3];
  float* out = (float*)d_out;                  // [smooth, commit, log_probs...]
  char* ws = (char*)d_ws;
  _Float16* ws_eh   = (_Float16*)ws;                    // 512*256*2 = 262144 B
  float*    ws_enorm= (float*)(ws + 262144);            // 512*4
  float*    ws_invn = (float*)(ws + 264192);            // 131072*4
  float*    ws_loss = (float*)(ws + 788480);            // 4*4
  float* logp = out + 2;

  hipLaunchKernelGGL(k_prep,    dim3(512),   dim3(256), 0, stream, ew, ws_eh, ws_enorm, ws_loss);
  hipLaunchKernelGGL(k_invnorm, dim3(16384), dim3(256), 0, stream, z, ws_invn);
  hipLaunchKernelGGL(k_masksum, dim3(64),    dim3(256), 0, stream, mask, ws_loss);
  hipLaunchKernelGGL(k_main,    dim3(2048),  dim3(256),
                     (size_t)(NPAD * EH_STRIDE * sizeof(_Float16)), stream,
                     z, mask, ws_eh, ws_enorm, ws_invn, step, logp, ws_loss);
  hipLaunchKernelGGL(k_smooth,  dim3(1024),  dim3(256), 0, stream, z, mask, ws_invn, ws_loss);
  hipLaunchKernelGGL(k_final,   dim3(1),     dim3(32),  0, stream, ws_loss, out);
}
